// Grid_SE3_20770461843835
// MI455X (gfx1250) — compile-verified, matrix operand order checked by reading, not run on hardware
//
#include <hip/hip_runtime.h>

#define NN 8192
#define NE 32768
#define ECH 4096

typedef _Float16 h16;
typedef __attribute__((ext_vector_type(16))) _Float16 v16h;
typedef __attribute__((ext_vector_type(8)))  float    v8f;

union V16H { v16h v; h16 h[16]; };
union V8F  { v8f  v; float f[8]; };

struct PairOff { long b1, b2, w1, w2; int ci, co, nJ; };
struct CGJob  { int l1, l2, l3, off; };
struct CGJobs { CGJob j[19]; };

// ============== CG coefficient generation (f64, once per call) ==============
__device__ double se3_fact(int n){ double r=1.0; for(int i=2;i<=n;++i) r*=(double)i; return r; }

__device__ double se3_su2cg(int j1,int m1,int j2,int m2,int j3,int m3){
  if (m3 != m1+m2) return 0.0;
  int vmin = 0;
  if (-j1+j2+m3 > vmin) vmin = -j1+j2+m3;
  if (-j1+m1   > vmin) vmin = -j1+m1;
  int vmax = j2+j3+m1;
  if (j3-j1+j2 < vmax) vmax = j3-j1+j2;
  if (j3+m3    < vmax) vmax = j3+m3;
  double C = sqrt((2.0*j3+1.0)*se3_fact(j3+j1-j2)*se3_fact(j3-j1+j2)*se3_fact(j1+j2-j3)*
                  se3_fact(j3+m3)*se3_fact(j3-m3)/
                  (se3_fact(j1+j2+j3+1)*se3_fact(j1-m1)*se3_fact(j1+m1)*
                   se3_fact(j2-m2)*se3_fact(j2+m2)));
  double S = 0.0;
  for (int v=vmin; v<=vmax; ++v){
    double sgn = ((v+j2+m2)&1)? -1.0: 1.0;
    S += sgn*se3_fact(j2+j3+m1-v)*se3_fact(j1-m1+v)/
         (se3_fact(v)*se3_fact(j3-j1+j2-v)*se3_fact(j3+m3-v)*se3_fact(v+j1-j2-m3));
  }
  return C*S;
}

__device__ void se3_ucb(int l, double* re, double* im){
  int n=2*l+1;
  for (int i=0;i<n*n;++i){ re[i]=0.0; im[i]=0.0; }
  double s = 0.70710678118654752440;
  for (int m=-l;m<=l;++m){
    int r=m+l;
    if (m<0){
      im[r*n + (m+l)] = s;
      double sg = ((-m)&1)? -1.0: 1.0;
      im[r*n + (-m+l)] = -sg*s;
    } else if (m==0){
      re[l*n+l] = 1.0;
    } else {
      re[r*n + (-m+l)] = s;
      double sg = (m&1)? -1.0: 1.0;
      re[r*n + (m+l)] = sg*s;
    }
  }
}

__global__ void se3_cg_kernel(CGJobs jobs, float* cg){
  int t = threadIdx.x;
  if (t >= 19) return;
  int l1=jobs.j[t].l1, l2=jobs.j[t].l2, l3=jobs.j[t].l3;
  int n1=2*l1+1, n2=2*l2+1, n3=2*l3+1;
  double U1re[25],U1im[25],U2re[81],U2im[81],U3re[25],U3im[25];
  se3_ucb(l1,U1re,U1im); se3_ucb(l2,U2re,U2im); se3_ucb(l3,U3re,U3im);
  double Rre[225], Rim[225];
  int tot = n1*n2*n3;
  for (int i=0;i<tot;++i){ Rre[i]=0.0; Rim[i]=0.0; }
  for (int a=0;a<n1;++a){
    int m1=a-l1;
    for (int b=0;b<n2;++b){
      int m2=b-l2; int m3=m1+m2;
      if (m3<-l3 || m3>l3) continue;
      int cC = m3+l3;
      double cgv = se3_su2cg(l1,m1,l2,m2,l3,m3);
      if (cgv==0.0) continue;
      for (int i=0;i<n1;++i){
        double u1r=U1re[i*n1+a], u1i=-U1im[i*n1+a];
        if (u1r==0.0 && u1i==0.0) continue;
        for (int jj=0;jj<n2;++jj){
          double u2r=U2re[jj*n2+b], u2i=-U2im[jj*n2+b];
          if (u2r==0.0 && u2i==0.0) continue;
          double pr=(u1r*u2r-u1i*u2i)*cgv;
          double pi=(u1r*u2i+u1i*u2r)*cgv;
          for (int k=0;k<n3;++k){
            double u3r=U3re[k*n3+cC], u3i=U3im[k*n3+cC];
            int idx=(i*n2+jj)*n3+k;
            Rre[idx] += pr*u3r - pi*u3i;
            Rim[idx] += pr*u3i + pi*u3r;
          }
        }
      }
    }
  }
  double nr=0.0, ni=0.0;
  for (int i=0;i<tot;++i){ nr+=Rre[i]*Rre[i]; ni+=Rim[i]*Rim[i]; }
  bool useRe = nr >= ni;
  float* outp = cg + jobs.j[t].off;
  for (int i=0;i<tot;++i) outp[i] = (float)(useRe? Rre[i] : Rim[i]);
}

// ============== elementwise helpers ==============
__global__ void se3_zero(float* p, long n){
  long i = (long)blockIdx.x*256 + threadIdx.x;
  if (i < n) p[i] = 0.f;
}
__global__ void se3_f2h(const float* a, h16* b, long n){
  long i = (long)blockIdx.x*256 + threadIdx.x;
  if (i < n) b[i] = (h16)a[i];
}

// ============== edge geometry: dist, rhat, Y0..Y4, ef(f16, padded 64) ==============
__global__ void se3_geom(const float* pos, const float* ef, const int* src, const int* dst,
                         float* Y, h16* efh){
  int e = blockIdx.x*256 + threadIdx.x;
  if (e >= NE) return;
  int s = src[e], d = dst[e];
  float rx = pos[d*3+0]-pos[s*3+0];
  float ry = pos[d*3+1]-pos[s*3+1];
  float rz = pos[d*3+2]-pos[s*3+2];
  float dist = sqrtf(rx*rx+ry*ry+rz*rz + 1e-8f);
  float x = rx/dist, y = ry/dist, z = rz/dist;
  float x2=x*x, y2=y*y, z2=z*z;
  float* Ye = Y + (long)e*25;
  Ye[0]=0.28209479f;
  Ye[1]=0.48860251f*y; Ye[2]=0.48860251f*z; Ye[3]=0.48860251f*x;
  Ye[4]=1.09254843f*x*y; Ye[5]=1.09254843f*y*z; Ye[6]=0.31539157f*(3.f*z2-1.f);
  Ye[7]=1.09254843f*x*z; Ye[8]=0.54627422f*(x2-y2);
  Ye[9]=0.59004359f*y*(3.f*x2-y2); Ye[10]=2.89061144f*x*y*z;
  Ye[11]=0.45704579f*y*(5.f*z2-1.f); Ye[12]=0.37317633f*z*(5.f*z2-3.f);
  Ye[13]=0.45704579f*x*(5.f*z2-1.f); Ye[14]=1.44530572f*z*(x2-y2);
  Ye[15]=0.59004359f*x*(x2-3.f*y2);
  Ye[16]=2.50334294f*x*y*(x2-y2); Ye[17]=1.77013077f*y*z*(3.f*x2-y2);
  Ye[18]=0.9461747f*x*y*(7.f*z2-1.f); Ye[19]=0.66904654f*y*z*(7.f*z2-3.f);
  Ye[20]=0.10578554f*(35.f*z2*z2-30.f*z2+3.f); Ye[21]=0.66904654f*x*z*(7.f*z2-3.f);
  Ye[22]=0.47308734f*(x2-y2)*(7.f*z2-1.f); Ye[23]=1.77013077f*x*z*(x2-3.f*y2);
  Ye[24]=0.62583573f*(x2*x2-6.f*x2*y2+y2*y2);
  h16* er = efh + (long)e*64;
  for (int c=0;c<32;++c) er[c] = (h16)ef[(long)e*32+c];
  er[32] = (h16)dist;
  for (int c=33;c<64;++c) er[c] = (h16)0.f;
}

// ============== WMMA tile loaders (CDNA5 16x16x32 f16 layout) ==============
__device__ inline v16h se3_ldA(const h16* A, int lda, int row0, int k0){
  int lane = threadIdx.x & 31;
  const h16* p = A + (long)(row0 + (lane & 15))*lda + k0 + ((lane>>4)<<3);
  V16H u;
#pragma unroll
  for (int i=0;i<8;++i) u.h[i] = p[i];
#pragma unroll
  for (int i=8;i<16;++i) u.h[i] = p[i+8];
  return u.v;
}
__device__ inline v16h se3_ldB(const h16* B, int ldb, int col0, int k0, int kmax){
  int lane = threadIdx.x & 31;
  int n = col0 + (lane & 15);
  int kb = k0 + ((lane>>4)<<3);
  V16H u;
#pragma unroll
  for (int i=0;i<16;++i){
    int k = kb + (i<8? i : i+8);
    u.h[i] = (k < kmax) ? B[(long)k*ldb + n] : (h16)0.f;
  }
  return u.v;
}

// radial stage 1: h = relu(ef @ w1 + b1)   [E x 64(pad)] @ [33 x 32] -> f16 [E x 32]
__global__ void se3_radial1(const h16* efh, const h16* w1, const float* b1, h16* hh){
  int mt = blockIdx.x, nt = blockIdx.y;
  v16h a0 = se3_ldA(efh, 64, mt*16, 0);
  v16h a1 = se3_ldA(efh, 64, mt*16, 32);
  v16h b0 = se3_ldB(w1, 32, nt*16, 0, 33);
  v16h b1v = se3_ldB(w1, 32, nt*16, 32, 33);
  v8f acc = {0.f,0.f,0.f,0.f,0.f,0.f,0.f,0.f};
  acc = __builtin_amdgcn_wmma_f32_16x16x32_f16(false, a0, false, b0,  (short)0, acc, false, false);
  acc = __builtin_amdgcn_wmma_f32_16x16x32_f16(false, a1, false, b1v, (short)0, acc, false, false);
  V8F c; c.v = acc;
  int lane = threadIdx.x & 31;
  int col = nt*16 + (lane & 15);
  int rbase = mt*16 + ((lane>>4)<<3);
#pragma unroll
  for (int v=0; v<8; ++v){
    float val = c.f[v] + b1[col];
    val = fmaxf(val, 0.f);
    hh[(long)(rbase+v)*32 + col] = (h16)val;
  }
}

// radial stage 2: R = h @ w2 + b2   [chunk x 32] @ [32 x Kout] -> f32
// One wave handles a 16x64 output strip: A tile loaded once, 4 WMMAs (Kout % 64 == 0).
__global__ void se3_radial2(const h16* hh, const h16* w2, const float* b2, float* R,
                            int e0, int Kout){
  int mt = blockIdx.x, ntq = blockIdx.y;
  v16h a = se3_ldA(hh, 32, e0 + mt*16, 0);
  int lane = threadIdx.x & 31;
  int rbase = mt*16 + ((lane>>4)<<3);
#pragma unroll
  for (int q=0; q<4; ++q){
    int n0 = ntq*64 + q*16;
    v16h b = se3_ldB(w2, Kout, n0, 0, 32);
    v8f acc = {0.f,0.f,0.f,0.f,0.f,0.f,0.f,0.f};
    acc = __builtin_amdgcn_wmma_f32_16x16x32_f16(false, a, false, b, (short)0, acc, false, false);
    V8F c; c.v = acc;
    int col = n0 + (lane & 15);
    float bias = b2[col];
#pragma unroll
    for (int v=0; v<8; ++v)
      R[(long)(rbase+v)*Kout + col] = c.f[v] + bias;
  }
}

// per-edge tensor-product conv consumer: out[e,a,o] += sum_{c,j} R[e,a,c,j]*t[e,c,o,j]
// R row staged global->LDS via CDNA5 async-to-LDS loads (ASYNCcnt path).
__global__ void se3_conv(const float* Y, const float* cgt, const float* R, const int* src,
                         const float* featIn, float* outBuf,
                         int e0, int di, int dd, int nJ, int ci, int co){
  __shared__ float sF[160];
  __shared__ float sB[125];
  __shared__ float sT[800];
  __shared__ float sR[1280];
  int e = e0 + blockIdx.x;
  int tid = threadIdx.x;
  int ni = 2*di+1, no = 2*dd+1;
  int Kout = co*ci*nJ;   // always a multiple of 64 -> async loop is EXEC-uniform
  int s = src[e];
  // async copy of R row into LDS (bypasses VGPRs, tracked by ASYNCcnt)
  const float* Rrow = R + (long)blockIdx.x*Kout;
  for (int i=tid; i<Kout; i+=64){
    unsigned ldsa = (unsigned)(size_t)(&sR[i]);
    unsigned goff = (unsigned)(i*4);
    asm volatile("global_load_async_to_lds_b32 %0, %1, %2"
                 :: "v"(ldsa), "v"(goff), "s"(Rrow) : "memory");
  }
  for (int i=tid; i<ci*ni; i+=64) sF[i] = featIn[(long)s*ci*ni + i];
  const float* Ye = Y + (long)e*25;
  int Jmin = (di>dd)? di-dd : dd-di;
  int nB = no*ni*nJ;
  for (int idx=tid; idx<nB; idx+=64){
    int j = idx % nJ; int i = (idx/nJ) % ni; int o = idx/(nJ*ni);
    int J = Jmin + j;
    int toff = 0;
    for (int t2=0; t2<j; ++t2) toff += ni*(2*(Jmin+t2)+1)*no;
    const float* tab = cgt + toff;
    int nm = 2*J+1;
    const float* Yj = Ye + J*J;
    float sum = 0.f;
    for (int m=0; m<nm; ++m) sum += Yj[m]*tab[(i*nm+m)*no + o];
    sB[(o*ni+i)*nJ + j] = sum;
  }
  asm volatile("s_wait_asynccnt 0x0" ::: "memory");
  __syncthreads();
  int nT = ci*no*nJ;
  for (int idx=tid; idx<nT; idx+=64){
    int j = idx % nJ; int o = (idx/nJ) % no; int c = idx/(nJ*no);
    float sum = 0.f;
    for (int i=0;i<ni;++i) sum += sB[(o*ni+i)*nJ + j]*sF[c*ni + i];
    sT[(c*no+o)*nJ + j] = sum;
  }
  __syncthreads();
  int nO = co*no;
  for (int idx=tid; idx<nO; idx+=64){
    int o = idx % no; int a = idx / no;
    float sum = 0.f;
    for (int c=0;c<ci;++c){
      const float* rp = sR + (a*ci+c)*nJ;
      const float* tp = sT + (c*no+o)*nJ;
      for (int j=0;j<nJ;++j) sum += rp[j]*tp[j];
    }
    outBuf[(long)e*nO + idx] += sum;
  }
}

// ============== attention machinery ==============
__global__ void se3_query(const float* feat, const float* par, long qoff, float* q, int nm){
  long idx = (long)blockIdx.x*256 + threadIdx.x;
  long tot = (long)NN*8*nm;
  if (idx >= tot) return;
  int m = (int)(idx % nm); int h = (int)((idx/nm) % 8); long n = idx/((long)nm*8);
  float sum = 0.f;
  for (int c=0;c<32;++c) sum += feat[(n*32+c)*nm + m]*par[qoff + c*8 + h];
  q[idx] = sum;
}

__global__ void se3_score(const float* q, const float* kb, const int* dst, float* sc, int nm){
  int idx = blockIdx.x*256 + threadIdx.x;
  if (idx >= NE*4) return;
  int h = idx & 3; int e = idx >> 2;
  long dn = dst[e];
  float sum = 0.f;
  for (int c2=0;c2<2;++c2){
    int ch = h*2 + c2;
    for (int m=0;m<nm;++m)
      sum += q[(dn*8+ch)*nm + m]*kb[((long)e*8+ch)*nm + m];
  }
  sc[idx] += sum;
}

__device__ inline unsigned se3_enc(float f){
  unsigned u = __float_as_uint(f);
  return (u & 0x80000000u) ? ~u : (u | 0x80000000u);
}

__global__ void se3_smax1(const float* sc, const int* dst, unsigned* menc, float scale){
  int idx = blockIdx.x*256 + threadIdx.x;
  if (idx >= NE*4) return;
  int h = idx & 3; int e = idx >> 2;
  atomicMax(&menc[dst[e]*4 + h], se3_enc(sc[idx]*scale));
}
__global__ void se3_smax2(const unsigned* menc, float* mval){
  int idx = blockIdx.x*256 + threadIdx.x;
  if (idx >= NN*4) return;
  unsigned u = menc[idx];
  float f = (u & 0x80000000u) ? __uint_as_float(u & 0x7FFFFFFFu) : __uint_as_float(~u);
  bool fin = (__float_as_uint(f) & 0x7F800000u) != 0x7F800000u;
  mval[idx] = fin ? f : 0.f;
}
__global__ void se3_smax3(const float* sc, const int* dst, const float* mval,
                          float* expv, float* z, float scale){
  int idx = blockIdx.x*256 + threadIdx.x;
  if (idx >= NE*4) return;
  int h = idx & 3; int e = idx >> 2;
  float v = expf(sc[idx]*scale - mval[dst[e]*4 + h]);
  expv[idx] = v;
  atomicAdd(&z[dst[e]*4 + h], v);
}
__global__ void se3_smax4(const float* expv, const int* dst, const float* z, float* a){
  int idx = blockIdx.x*256 + threadIdx.x;
  if (idx >= NE*4) return;
  int h = idx & 3; int e = idx >> 2;
  a[idx] = expv[idx]/(z[dst[e]*4 + h] + 1e-9f);
}

__global__ void se3_agg(const float* a, const float* vb, const int* dst, float* agg, int nm){
  long idx = (long)blockIdx.x*256 + threadIdx.x;
  long tot = (long)NE*8*nm;
  if (idx >= tot) return;
  int m = (int)(idx % nm); int c = (int)((idx/nm) % 8); long e = idx/((long)nm*8);
  atomicAdd(&agg[((long)dst[e]*8 + c)*nm + m], a[e*4 + (c>>1)]*vb[idx]);
}

__global__ void se3_proj(const float* agg, const float* featd, const float* par, long poff,
                         float* outF, int nm){
  long idx = (long)blockIdx.x*256 + threadIdx.x;
  long tot = (long)NN*32*nm;
  if (idx >= tot) return;
  int m = (int)(idx % nm); int h = (int)((idx/nm) % 32); long n = idx/((long)nm*32);
  float sum = 0.f;
  for (int c=0;c<8;++c) sum += agg[(n*8+c)*nm + m]*par[poff + c*32 + h];
  if (featd)
    for (int c=0;c<32;++c) sum += featd[(n*32+c)*nm + m]*par[poff + (8+c)*32 + h];
  outF[idx] = sum;
}

__global__ void se3_norm(float* feat, const float* par, long goff, long boff, int nm){
  int n = blockIdx.x, c = threadIdx.x;   // wave32: 32 channels == one wave
  float xs[5]; float ss = 0.f;
  for (int m=0;m<nm;++m){ xs[m] = feat[((long)n*32+c)*nm + m]; ss += xs[m]*xs[m]; }
  float nrm = sqrtf(ss + 1e-12f);
  float mu = nrm;
  for (int o=16;o>0;o>>=1) mu += __shfl_xor(mu, o, 32);
  mu *= (1.f/32.f);
  float dv = nrm - mu;
  float var = dv*dv;
  for (int o=16;o>0;o>>=1) var += __shfl_xor(var, o, 32);
  var *= (1.f/32.f);
  float hv = dv*rsqrtf(var + 1e-5f)*par[goff + c] + par[boff + c];
  float fac = fmaxf(hv, 0.f)/nrm;
  for (int m=0;m<nm;++m) feat[((long)n*32+c)*nm + m] = xs[m]*fac;
}

__global__ void se3_self(const float* f0, const float* par, long wso, float* h0){
  int idx = blockIdx.x*256 + threadIdx.x;
  if (idx >= NN*32) return;
  int h = idx & 31; int n = idx >> 5;
  float sum = 0.f;
  for (int c=0;c<32;++c) sum += f0[(long)n*32 + c]*par[wso + c*32 + h];
  h0[idx] = sum;
}
__global__ void se3_scat(const float* msg, const int* dst, float* h0){
  int idx = blockIdx.x*256 + threadIdx.x;
  if (idx >= NE*32) return;
  int c = idx & 31; int e = idx >> 5;
  atomicAdd(&h0[(long)dst[e]*32 + c], msg[idx]);
}
__global__ void se3_copy(const float* h0, float* outp){
  int idx = blockIdx.x*256 + threadIdx.x;
  if (idx < NN*32) outp[idx] = h0[idx];
}
__global__ void se3_cs(const float* h0, const float* par, long wcoff, float* outp){
  int idx = blockIdx.x*256 + threadIdx.x;
  if (idx >= NN*15) return;
  int n = idx/15; int t = idx%15;
  float sum = 0.f;
  for (int c=0;c<32;++c) sum += h0[(long)n*32 + c]*par[wcoff + c*15 + t];
  outp[NN*32 + idx] = sum;
}

// =====================================================================
extern "C" void kernel_launch(void* const* d_in, const int* in_sizes, int n_in,
                              void* d_out, int out_size, void* d_ws, size_t ws_size,
                              hipStream_t stream){
  (void)in_sizes; (void)n_in; (void)out_size; (void)ws_size;
  const float* par       = (const float*)d_in[0];
  const float* node_feat = (const float*)d_in[1];
  const float* edge_feat = (const float*)d_in[2];
  const float* pos       = (const float*)d_in[3];
  const int*   srcI      = (const int*)d_in[4];
  const int*   dstI      = (const int*)d_in[5];
  float* outp = (float*)d_out;

  // ---- param offsets (jax pytree flatten order: dict keys sorted) ----
  long off = 0;
  auto pairAt = [&](int ci,int co,int nJ){
    PairOff p; p.ci=ci; p.co=co; p.nJ=nJ;
    p.b1 = off; off += 32;
    p.b2 = off; off += (long)co*ci*nJ;
    p.w1 = off; off += 33*32;
    p.w2 = off; off += 32L*co*ci*nJ;
    return p;
  };
  PairOff fin[3];
  for (int di=0;di<3;++di) fin[di] = pairAt(32,32,1);          // final/00,10,20
  PairOff l0key = pairAt(32,8,1);                               // layer0/key/00
  long l0proj[3]; l0proj[0]=off; off+=40*32; l0proj[1]=off; off+=8*32; l0proj[2]=off; off+=8*32;
  long l0query = off; off += 32*8;
  PairOff l0val[3];
  for (int dd=0;dd<3;++dd) l0val[dd] = pairAt(32,8,1);          // layer0/val/00,01,02
  PairOff l1key[3][3];
  for (int di=0;di<3;++di) for (int dd=0;dd<3;++dd){
    int mn = di<dd? di:dd;
    l1key[di][dd] = pairAt(32,8,2*mn+1);                        // layer1/key/00..22
  }
  long l1proj[3];  for (int d=0;d<3;++d){ l1proj[d]=off; off+=40*32; }
  long l1query[3]; for (int d=0;d<3;++d){ l1query[d]=off; off+=32*8; }
  PairOff l1val[3][3];
  for (int di=0;di<3;++di) for (int dd=0;dd<3;++dd){
    int mn = di<dd? di:dd;
    l1val[di][dd] = pairAt(32,8,2*mn+1);                        // layer1/val/00..22
  }
  long n0b[3], n0g[3], n1b[3], n1g[3];
  for (int d=0;d<3;++d){ n0b[d]=off; off+=32; n0g[d]=off; off+=32; }
  for (int d=0;d<3;++d){ n1b[d]=off; off+=32; n1g[d]=off; off+=32; }
  long wso = off; off += 32*32;
  long wcoff = off; off += 32*15;
  long nPar = off;   // 491936

  // ---- CG table layout ----
  CGJobs jobs; int cgPairBase[3][3]; int cgoff = 0; int tj = 0;
  for (int di=0;di<3;++di) for (int dd=0;dd<3;++dd){
    cgPairBase[di][dd] = cgoff;
    int Jmin = di>dd? di-dd : dd-di;
    for (int J=Jmin; J<=di+dd; ++J){
      jobs.j[tj].l1=di; jobs.j[tj].l2=J; jobs.j[tj].l3=dd; jobs.j[tj].off=cgoff; ++tj;
      cgoff += (2*di+1)*(2*J+1)*(2*dd+1);
    }
  }

  // ---- workspace layout ----
  size_t wo = 0;
  auto alloc = [&](size_t b){ size_t r = wo; wo += (b + 255) & ~(size_t)255; return r; };
  char* ws = (char*)d_ws;
  float* cgf  = (float*)(ws + alloc(sizeof(float)*(size_t)cgoff));
  float* Yb   = (float*)(ws + alloc(sizeof(float)*(size_t)NE*25));
  h16*   efh  = (h16*)  (ws + alloc(2ull*NE*64));
  h16*   parh = (h16*)  (ws + alloc(2ull*(size_t)nPar));
  h16*   hh   = (h16*)  (ws + alloc(2ull*NE*32));
  float* Rb   = (float*)(ws + alloc(4ull*ECH*1280));
  float *qb[3], *kbb[3], *vbb[3], *aggb[3], *fA[3], *fB[3];
  for (int d=0;d<3;++d){
    int nm = 2*d+1;
    qb[d]   = (float*)(ws + alloc(4ull*NN*8*nm));
    kbb[d]  = (float*)(ws + alloc(4ull*NE*8*nm));
    vbb[d]  = (float*)(ws + alloc(4ull*NE*8*nm));
    aggb[d] = (float*)(ws + alloc(4ull*NN*8*nm));
    fA[d]   = (float*)(ws + alloc(4ull*NN*32*nm));
    fB[d]   = (float*)(ws + alloc(4ull*NN*32*nm));
  }
  float* scb  = (float*)(ws + alloc(4ull*NE*4));
  float* expv = (float*)(ws + alloc(4ull*NE*4));
  float* ab   = (float*)(ws + alloc(4ull*NE*4));
  unsigned* menc = (unsigned*)(ws + alloc(4ull*NN*4));
  float* mval = (float*)(ws + alloc(4ull*NN*4));
  float* zb   = (float*)(ws + alloc(4ull*NN*4));
  float* msg  = (float*)(ws + alloc(4ull*NE*32));
  float* h0b  = (float*)(ws + alloc(4ull*NN*32));

  auto blocks = [](long n){ return dim3((unsigned)((n + 255)/256)); };
  auto zero = [&](float* p, long n){ se3_zero<<<blocks(n), 256, 0, stream>>>(p, n); };

  // ---- init ----
  se3_cg_kernel<<<1, 32, 0, stream>>>(jobs, cgf);
  se3_f2h<<<blocks(nPar), 256, 0, stream>>>(par, parh, nPar);
  se3_geom<<<blocks(NE), 256, 0, stream>>>(pos, edge_feat, srcI, dstI, Yb, efh);

  auto runPair = [&](const PairOff& P, int di, int dd, const float* featIn, float* outB){
    int Kout = P.co*P.ci*P.nJ;
    se3_radial1<<<dim3(NE/16, 2), 32, 0, stream>>>(efh, parh + P.w1, par + P.b1, hh);
    for (int e0=0; e0<NE; e0+=ECH){
      se3_radial2<<<dim3(ECH/16, Kout/64), 32, 0, stream>>>(hh, parh + P.w2, par + P.b2, Rb, e0, Kout);
      se3_conv<<<dim3(ECH), 64, 0, stream>>>(Yb, cgf + cgPairBase[di][dd], Rb, srcI,
                                             featIn, outB, e0, di, dd, P.nJ, P.ci, P.co);
    }
  };

  auto softmax = [&](float scale){
    zero((float*)menc, NN*4); zero(zb, NN*4);
    se3_smax1<<<blocks(NE*4), 256, 0, stream>>>(scb, dstI, menc, scale);
    se3_smax2<<<blocks(NN*4), 256, 0, stream>>>(menc, mval);
    se3_smax3<<<blocks(NE*4), 256, 0, stream>>>(scb, dstI, mval, expv, zb, scale);
    se3_smax4<<<blocks(NE*4), 256, 0, stream>>>(expv, dstI, zb, ab);
  };

  // ================= layer 0 (in_degs = {0}) =================
  for (int dd=0;dd<3;++dd){
    zero(vbb[dd], (long)NE*8*(2*dd+1));
    runPair(l0val[dd], 0, dd, node_feat, vbb[dd]);
  }
  zero(kbb[0], (long)NE*8);
  runPair(l0key, 0, 0, node_feat, kbb[0]);
  se3_query<<<blocks((long)NN*8), 256, 0, stream>>>(node_feat, par, l0query, qb[0], 1);
  zero(scb, NE*4);
  se3_score<<<blocks(NE*4), 256, 0, stream>>>(qb[0], kbb[0], dstI, scb, 1);
  softmax(0.70710678f);                         // 1/sqrt(8/4)
  for (int d=0;d<3;++d){
    int nm = 2*d+1;
    zero(aggb[d], (long)NN*8*nm);
    se3_agg<<<blocks((long)NE*8*nm), 256, 0, stream>>>(ab, vbb[d], dstI, aggb[d], nm);
    se3_proj<<<blocks((long)NN*32*nm), 256, 0, stream>>>(aggb[d],
        (d==0 ? node_feat : (const float*)nullptr), par, l0proj[d], fA[d], nm);
    se3_norm<<<dim3(NN), 32, 0, stream>>>(fA[d], par, n0g[d], n0b[d], nm);
  }

  // ================= layer 1 (in_degs = {0,1,2}) =================
  for (int dd=0;dd<3;++dd){
    zero(vbb[dd], (long)NE*8*(2*dd+1));
    zero(kbb[dd], (long)NE*8*(2*dd+1));
    for (int di=0;di<3;++di){
      runPair(l1val[di][dd], di, dd, fA[di], vbb[dd]);
      runPair(l1key[di][dd], di, dd, fA[di], kbb[dd]);
    }
  }
  zero(scb, NE*4);
  for (int d=0;d<3;++d){
    int nm = 2*d+1;
    se3_query<<<blocks((long)NN*8*nm), 256, 0, stream>>>(fA[d], par, l1query[d], qb[d], nm);
    se3_score<<<blocks(NE*4), 256, 0, stream>>>(qb[d], kbb[d], dstI, scb, nm);
  }
  softmax(0.23570226f);                         // 1/sqrt(72/4)
  for (int d=0;d<3;++d){
    int nm = 2*d+1;
    zero(aggb[d], (long)NN*8*nm);
    se3_agg<<<blocks((long)NE*8*nm), 256, 0, stream>>>(ab, vbb[d], dstI, aggb[d], nm);
    se3_proj<<<blocks((long)NN*32*nm), 256, 0, stream>>>(aggb[d], fA[d], par, l1proj[d], fB[d], nm);
    se3_norm<<<dim3(NN), 32, 0, stream>>>(fB[d], par, n1g[d], n1b[d], nm);
  }

  // ================= final conv + readout =================
  zero(msg, (long)NE*32);
  for (int di=0;di<3;++di) runPair(fin[di], di, 0, fB[di], msg);
  se3_self<<<blocks((long)NN*32), 256, 0, stream>>>(fB[0], par, wso, h0b);
  se3_scat<<<blocks((long)NE*32), 256, 0, stream>>>(msg, dstI, h0b);
  se3_copy<<<blocks((long)NN*32), 256, 0, stream>>>(h0b, outp);
  se3_cs<<<blocks((long)NN*15), 256, 0, stream>>>(h0b, par, wcoff, outp);
}
